// MotionNetwork_71665824301577
// MI455X (gfx1250) — matrix operand order repaired and checked by reading, not compiled, not run on hardware
//
#include <hip/hip_runtime.h>
#include <math.h>

typedef _Float16 f16;
typedef __attribute__((ext_vector_type(16))) _Float16 v16h;
typedef __attribute__((ext_vector_type(8)))  _Float16 v8h;
typedef __attribute__((ext_vector_type(8)))  float    v8f;

#define PRIME 2654435761u
#define TBL   131072
#define TMASK 131071u

// ---- workspace layout ----
// float region (at byte 0):
//   [0..31]  enc_a1
//   [32..37] fused
//   [38]     amb_eye
//   [40..51] RES[12]
// f16 region (at byte 256), element offsets:
enum : int { AUD0 = 0, AUD1 = 2048, SIG0 = 3072, SIG1 = 9216, SIG2 = 13312 };
// AUD0: audch.w0 as [32][64]  (K 36 -> 64, zero pad)
// AUD1: audch.w1 as [32][32]
// SIG0: sigma.w0 as [64][96]  (K 74 -> 96, zero pad)
// SIG1: sigma.w1 as [64][64]
// SIG2: sigma.w2 as [16][64]  (N 11 -> 16, zero pad rows)

struct PtrPack { const float* q[45]; };

__device__ __forceinline__ float leaky(float x) { return x > 0.f ? x : 0.02f * x; }

// ---------------------------------------------------------------------------
// Prep kernel: batch-independent audio / attention / gate path + weight f16
// conversion + RES table.  One block of 256 threads.
// ---------------------------------------------------------------------------
__global__ __launch_bounds__(256) void motion_prep(PtrPack pp, float* wsF, f16* wsH) {
    const int tid = threadIdx.x;
    const float* a = pp.q[1];
    const float* e = pp.q[2];

    // ---- weight conversion to padded f16 ----
    { const float* w = pp.q[32];                      // audch.w0 (32,36)
      for (int i = tid; i < 32 * 64; i += 256) { int oc = i >> 6, k = i & 63;
          wsH[AUD0 + i] = (f16)(k < 36 ? w[oc * 36 + k] : 0.f); } }
    { const float* w = pp.q[33];                      // audch.w1 (32,32)
      for (int i = tid; i < 32 * 32; i += 256) wsH[AUD1 + i] = (f16)w[i]; }
    { const float* w = pp.q[42];                      // sigma.w0 (64,74)
      for (int i = tid; i < 64 * 96; i += 256) { int o = i / 96, k = i % 96;
          wsH[SIG0 + i] = (f16)(k < 74 ? w[o * 74 + k] : 0.f); } }
    { const float* w = pp.q[43];                      // sigma.w1 (64,64)
      for (int i = tid; i < 64 * 64; i += 256) wsH[SIG1 + i] = (f16)w[i]; }
    { const float* w = pp.q[44];                      // sigma.w2 (11,64)
      for (int i = tid; i < 16 * 64; i += 256) { int o = i >> 6, k = i & 63;
          wsH[SIG2 + i] = (f16)(o < 11 ? w[o * 64 + k] : 0.f); } }

    // ---- RES table ----
    if (tid < 12) wsF[40 + tid] = ceilf(16.f * expf(logf(2.4f) * (float)tid / 11.f));

    // ---- audio conv chain ----
    __shared__ float s1[8 * 32 * 8];    // [b][32][8]
    __shared__ float s2[8 * 32 * 4];    // [b][32][4]
    __shared__ float s3[8 * 64 * 2];    // [b][64][2]
    __shared__ float y4[8 * 64];        // [b][64]
    __shared__ float yf1[8 * 64];       // [b][64]
    __shared__ float yy[8 * 32];        // [b][32]  (f2 out)
    __shared__ float z1[16 * 8], z2[8 * 8], z3[4 * 8], z4[2 * 8], z5[8];
    __shared__ float attw[8], encAs[32];

    { const float* w = pp.q[6]; const float* bb = pp.q[7];     // c1 (32,29,3) s2 p1
      for (int t = tid; t < 2048; t += 256) {
          int b = t >> 8, oc = (t >> 3) & 31, ol = t & 7;
          float s = bb[oc];
          for (int ic = 0; ic < 29; ++ic)
              for (int k = 0; k < 3; ++k) {
                  int il = ol * 2 - 1 + k;
                  if (il >= 0 && il < 16) s += w[oc * 87 + ic * 3 + k] * a[b * 464 + ic * 16 + il];
              }
          s1[t] = leaky(s);
      } }
    __syncthreads();
    { const float* w = pp.q[8]; const float* bb = pp.q[9];     // c2 (32,32,3)
      for (int t = tid; t < 1024; t += 256) {
          int b = t >> 7, oc = (t >> 2) & 31, ol = t & 3;
          float s = bb[oc];
          for (int ic = 0; ic < 32; ++ic)
              for (int k = 0; k < 3; ++k) {
                  int il = ol * 2 - 1 + k;
                  if (il >= 0 && il < 8) s += w[oc * 96 + ic * 3 + k] * s1[b * 256 + ic * 8 + il];
              }
          s2[b * 128 + oc * 4 + ol] = leaky(s);
      } }
    __syncthreads();
    { const float* w = pp.q[10]; const float* bb = pp.q[11];   // c3 (64,32,3)
      for (int t = tid; t < 1024; t += 256) {
          int b = t >> 7, oc = (t >> 1) & 63, ol = t & 1;
          float s = bb[oc];
          for (int ic = 0; ic < 32; ++ic)
              for (int k = 0; k < 3; ++k) {
                  int il = ol * 2 - 1 + k;
                  if (il >= 0 && il < 4) s += w[oc * 96 + ic * 3 + k] * s2[b * 128 + ic * 4 + il];
              }
          s3[b * 128 + oc * 2 + ol] = leaky(s);
      } }
    __syncthreads();
    { const float* w = pp.q[12]; const float* bb = pp.q[13];   // c4 (64,64,3)
      for (int t = tid; t < 512; t += 256) {
          int b = t >> 6, oc = t & 63;
          float s = bb[oc];
          for (int ic = 0; ic < 64; ++ic)
              for (int k = 0; k < 3; ++k) {
                  int il = -1 + k;
                  if (il >= 0 && il < 2) s += w[oc * 192 + ic * 3 + k] * s3[b * 128 + ic * 2 + il];
              }
          y4[t] = leaky(s);
      } }
    __syncthreads();
    { const float* w = pp.q[14]; const float* bb = pp.q[15];   // f1 (64,64)
      for (int t = tid; t < 512; t += 256) {
          int b = t >> 6, o = t & 63;
          float s = bb[o];
          for (int i = 0; i < 64; ++i) s += w[o * 64 + i] * y4[b * 64 + i];
          yf1[t] = leaky(s);
      } }
    __syncthreads();
    { const float* w = pp.q[16]; const float* bb = pp.q[17];   // f2 (32,64), no act
      for (int t = tid; t < 256; t += 256) {
          int b = t >> 5, o = t & 31;
          float s = bb[o];
          for (int i = 0; i < 64; ++i) s += w[o * 64 + i] * yf1[b * 64 + i];
          yy[t] = s;
      } }
    __syncthreads();
    // ---- attention convs (z0[c][t] = yy[t*32+c], len 8, stride 1, pad 1) ----
    { const float* w = pp.q[18]; const float* bb = pp.q[19];   // (16,32,3)
      for (int t = tid; t < 128; t += 256) {
          int oc = t >> 3, ot = t & 7; float s = bb[oc];
          for (int ic = 0; ic < 32; ++ic)
              for (int k = 0; k < 3; ++k) { int il = ot - 1 + k;
                  if (il >= 0 && il < 8) s += w[oc * 96 + ic * 3 + k] * yy[il * 32 + ic]; }
          z1[t] = leaky(s);
      } }
    __syncthreads();
    { const float* w = pp.q[20]; const float* bb = pp.q[21];   // (8,16,3)
      for (int t = tid; t < 64; t += 256) {
          int oc = t >> 3, ot = t & 7; float s = bb[oc];
          for (int ic = 0; ic < 16; ++ic)
              for (int k = 0; k < 3; ++k) { int il = ot - 1 + k;
                  if (il >= 0 && il < 8) s += w[oc * 48 + ic * 3 + k] * z1[ic * 8 + il]; }
          z2[t] = leaky(s);
      } }
    __syncthreads();
    { const float* w = pp.q[22]; const float* bb = pp.q[23];   // (4,8,3)
      for (int t = tid; t < 32; t += 256) {
          int oc = t >> 3, ot = t & 7; float s = bb[oc];
          for (int ic = 0; ic < 8; ++ic)
              for (int k = 0; k < 3; ++k) { int il = ot - 1 + k;
                  if (il >= 0 && il < 8) s += w[oc * 24 + ic * 3 + k] * z2[ic * 8 + il]; }
          z3[t] = leaky(s);
      } }
    __syncthreads();
    { const float* w = pp.q[24]; const float* bb = pp.q[25];   // (2,4,3)
      for (int t = tid; t < 16; t += 256) {
          int oc = t >> 3, ot = t & 7; float s = bb[oc];
          for (int ic = 0; ic < 4; ++ic)
              for (int k = 0; k < 3; ++k) { int il = ot - 1 + k;
                  if (il >= 0 && il < 8) s += w[oc * 12 + ic * 3 + k] * z3[ic * 8 + il]; }
          z4[t] = leaky(s);
      } }
    __syncthreads();
    { const float* w = pp.q[26]; const float* bb = pp.q[27];   // (1,2,3)
      for (int t = tid; t < 8; t += 256) {
          float s = bb[0];
          for (int ic = 0; ic < 2; ++ic)
              for (int k = 0; k < 3; ++k) { int il = t - 1 + k;
                  if (il >= 0 && il < 8) s += w[ic * 3 + k] * z4[ic * 8 + il]; }
          z5[t] = leaky(s);
      } }
    __syncthreads();
    if (tid == 0) {   // logits + softmax
        const float* lw = pp.q[28]; const float* lb = pp.q[29];
        float lg[8], m = -1e30f;
        for (int j = 0; j < 8; ++j) { float s = lb[j];
            for (int i = 0; i < 8; ++i) s += lw[j * 8 + i] * z5[i];
            lg[j] = s; m = fmaxf(m, s); }
        float den = 0.f;
        for (int j = 0; j < 8; ++j) { lg[j] = expf(lg[j] - m); den += lg[j]; }
        for (int j = 0; j < 8; ++j) attw[j] = lg[j] / den;
    }
    __syncthreads();
    if (tid < 32) {   // enc_a1
        float s = 0.f;
        for (int j = 0; j < 8; ++j) s += attw[j] * yy[j * 32 + tid];
        encAs[tid] = s; wsF[tid] = s;
    }
    __syncthreads();
    if (tid == 0) {   // gate path constants
        const float* xw0 = pp.q[30]; const float* xw1 = pp.q[31];
        float h16[16], ee[6];
        for (int j = 0; j < 16; ++j) { float s = 0.f;
            for (int i = 0; i < 5; ++i) s += xw0[j * 5 + i] * e[i];
            h16[j] = fmaxf(s, 0.f); }
        for (int o = 0; o < 5; ++o) { float s = 0.f;
            for (int j = 0; j < 16; ++j) s += xw1[o * 16 + j] * h16[j];
            ee[o] = s; }
        ee[5] = e[5];
        const float* vw = pp.q[34]; const float* vb = pp.q[35];
        const float* aw = pp.q[36]; const float* ab = pp.q[37];
        const float* tw = pp.q[38]; const float* tb = pp.q[39];
        float vis[6], aud[6];
        for (int v = 0; v < 6; ++v) { float s = vb[v];
            for (int i = 0; i < 6; ++i) s += vw[v * 6 + i] * ee[i];
            vis[v] = fmaxf(s, 0.f); }
        for (int v = 0; v < 6; ++v) { float s = ab[v];
            for (int i = 0; i < 32; ++i) s += aw[v * 32 + i] * encAs[i];
            aud[v] = fmaxf(s, 0.f); }
        float ts = tb[0];
        for (int i = 0; i < 6; ++i) ts += tw[i] * vis[i];
        for (int i = 0; i < 6; ++i) ts += tw[6 + i] * aud[i];
        float eye = 1.f / (1.f + expf(-fmaxf(ts, 0.f)));
        const float* pw = pp.q[40]; const float* pb = pp.q[41];
        for (int v = 0; v < 6; ++v) { float s = pb[v];
            for (int i = 0; i < 32; ++i) s += pw[v * 32 + i] * encAs[i];
            wsF[32 + v] = eye * s + ee[v]; }
        wsF[38] = eye;   // amb_eye = |eye| = eye
    }
}

// ---------------------------------------------------------------------------
// WMMA helpers (ISA 16-bit layouts)
// ---------------------------------------------------------------------------
__device__ __forceinline__ v8f wmma_f16(v16h a, v16h b, v8f c) {
    return __builtin_amdgcn_wmma_f32_16x16x32_f16(false, a, false, b, (short)0, c, false, false);
}

// A: 16x32 tile from row-major [rows][stride] f16; runs at kc+half*8 and kc+16+half*8
__device__ __forceinline__ v16h loadA(const f16* base, int rowBase, int stride, int kc) {
    int lane = threadIdx.x & 31;
    int row  = lane & 15, half = lane >> 4;
    const f16* p = base + (rowBase + row) * stride + kc + half * 8;
    v8h lo = *(const v8h*)p;
    v8h hi = *(const v8h*)(p + 16);
    v16h r;
    #pragma unroll
    for (int i = 0; i < 8; ++i) { r[i] = lo[i]; r[i + 8] = hi[i]; }
    return r;
}

// B: 32x16 tile from W[n][kpad] f16; runs at kc+half*8 and kc+16+half*8
__device__ __forceinline__ v16h loadB(const f16* w, int ntile, int kpad, int kc) {
    int lane = threadIdx.x & 31;
    int n = (lane & 15) + ntile * 16, half = lane >> 4;
    const f16* p = w + n * kpad + kc + half * 8;
    v8h lo = *(const v8h*)p;
    v8h hi = *(const v8h*)(p + 16);
    v16h r;
    #pragma unroll
    for (int i = 0; i < 8; ++i) { r[i] = lo[i]; r[i + 8] = hi[i]; }
    return r;
}

// ---------------------------------------------------------------------------
// Main kernel: 128 points / block, 8 waves x 16 rows.
// ---------------------------------------------------------------------------
__global__ __launch_bounds__(256) void motion_main(
        const float* __restrict__ x,
        const float* __restrict__ txy, const float* __restrict__ tyz,
        const float* __restrict__ txz,
        const float* __restrict__ wsF, const f16* __restrict__ wsH,
        float* __restrict__ out, int N) {
    __shared__ __align__(16) f16 Xs[128 * 96];   // activations [row][96]
    __shared__ __align__(16) f16 Hs[128 * 64];   // L0 out [row][64]; also audch H1 [row][32]
    __shared__ __align__(16) f16 H2s[128 * 64];  // L1 out [row][64]

    const int tid = threadIdx.x;
    const int blockBase = blockIdx.x * 128;

    // ---- init cols 36..95: zero + fused constant ----
    if (tid < 128) {
        #pragma unroll
        for (int k = 36; k < 96; ++k) Xs[tid * 96 + k] = (f16)0.f;
        for (int j = 0; j < 6; ++j) Xs[tid * 96 + 68 + j] = (f16)wsF[32 + j];
    }

    // ---- hash encoding: 128 points x 36 (plane,level) features ----
    for (int t = tid; t < 128 * 36; t += 256) {
        int p = t & 127, f = t >> 7;
        int plane = f / 12, lvl = f - plane * 12;
        int pg = blockBase + p; if (pg >= N) pg = N - 1;
        float c0, c1;
        if (plane == 0)      { c0 = x[3 * pg + 0]; c1 = x[3 * pg + 1]; }
        else if (plane == 1) { c0 = x[3 * pg + 1]; c1 = x[3 * pg + 2]; }
        else                 { c0 = x[3 * pg + 0]; c1 = x[3 * pg + 2]; }
        float u = fminf(fmaxf((c0 + 0.15f) / 0.3f, 0.f), 1.f);
        float v = fminf(fmaxf((c1 + 0.15f) / 0.3f, 0.f), 1.f);
        float res = wsF[40 + lvl];
        float px = u * res, py = v * res;
        float fx = px - floorf(px), fy = py - floorf(py);
        unsigned i0 = (unsigned)floorf(px), i1 = (unsigned)floorf(py);
        const float* tab = (plane == 0 ? txy : plane == 1 ? tyz : txz) + lvl * TBL;
        float c00 = tab[( i0        ^ ( i1      * PRIME)) & TMASK];
        float c10 = tab[((i0 + 1u)  ^ ( i1      * PRIME)) & TMASK];
        float c01 = tab[( i0        ^ ((i1 + 1u) * PRIME)) & TMASK];
        float c11 = tab[((i0 + 1u)  ^ ((i1 + 1u) * PRIME)) & TMASK];
        float feat = (1.f - fx) * (1.f - fy) * c00 + fx * (1.f - fy) * c10
                   + (1.f - fx) * fy * c01 + fx * fy * c11;
        Xs[p * 96 + plane * 12 + lvl] = (f16)feat;
    }
    __syncthreads();

    const int wave = tid >> 5, lane = tid & 31;
    const int rowBase = wave * 16;
    const int nIdx = lane & 15, half = lane >> 4;
    const float ambEye = wsF[38];

    // ---- aud_ch_att layer 1: relu(X[:, :64] @ AUD0^T) -> H1 [row][32] ----
    f16* H1 = Hs; // stride 32
    #pragma unroll
    for (int ntile = 0; ntile < 2; ++ntile) {
        v8f acc = {};
        acc = wmma_f16(loadA(Xs, rowBase, 96, 0),  loadB(wsH + AUD0, ntile, 64, 0),  acc);
        acc = wmma_f16(loadA(Xs, rowBase, 96, 32), loadB(wsH + AUD0, ntile, 64, 32), acc);
        #pragma unroll
        for (int r = 0; r < 8; ++r) {
            int m = r + half * 8;
            float v = acc[r];
            H1[(rowBase + m) * 32 + ntile * 16 + nIdx] = (f16)(v > 0.f ? v : 0.f);
        }
    }

    // ---- aud_ch_att layer 2 + enc_w + amb_aud ----
    float ssq[8];
    #pragma unroll
    for (int r = 0; r < 8; ++r) ssq[r] = 0.f;
    #pragma unroll
    for (int ntile = 0; ntile < 2; ++ntile) {
        v8f acc = {};
        acc = wmma_f16(loadA(H1, rowBase, 32, 0), loadB(wsH + AUD1, ntile, 32, 0), acc);
        int col = ntile * 16 + nIdx;
        float ea = wsF[col];                // enc_a1[col]
        #pragma unroll
        for (int r = 0; r < 8; ++r) {
            float v = acc[r];
            ssq[r] += v * v;
            Xs[(rowBase + r + half * 8) * 96 + 36 + col] = (f16)(ea * v);
        }
    }
    // reduce sum-of-squares across the 16 lanes of each half (rows r / r+8)
    #pragma unroll
    for (int step = 1; step < 16; step <<= 1)
        #pragma unroll
        for (int r = 0; r < 8; ++r) ssq[r] += __shfl_xor(ssq[r], step, 16);
    if (nIdx == 0) {
        #pragma unroll
        for (int r = 0; r < 8; ++r) {
            int p = blockBase + rowBase + r + half * 8;
            if (p < N) {
                out[p * 13 + 11] = sqrtf(ssq[r]);
                out[p * 13 + 12] = ambEye;
            }
        }
    }
    __syncthreads();   // H1 dead; Hs about to be reused block-wide

    // ---- sigma layer 0: relu(X[:, :96] @ SIG0^T) -> Hs [row][64] ----
    #pragma unroll
    for (int ntile = 0; ntile < 4; ++ntile) {
        v8f acc = {};
        acc = wmma_f16(loadA(Xs, rowBase, 96, 0),  loadB(wsH + SIG0, ntile, 96, 0),  acc);
        acc = wmma_f16(loadA(Xs, rowBase, 96, 32), loadB(wsH + SIG0, ntile, 96, 32), acc);
        acc = wmma_f16(loadA(Xs, rowBase, 96, 64), loadB(wsH + SIG0, ntile, 96, 64), acc);
        #pragma unroll
        for (int r = 0; r < 8; ++r) {
            int m = r + half * 8;
            float v = acc[r];
            Hs[(rowBase + m) * 64 + ntile * 16 + nIdx] = (f16)(v > 0.f ? v : 0.f);
        }
    }
    // ---- sigma layer 1: relu(H @ SIG1^T) -> H2 [row][64]  (own rows only) ----
    #pragma unroll
    for (int ntile = 0; ntile < 4; ++ntile) {
        v8f acc = {};
        acc = wmma_f16(loadA(Hs, rowBase, 64, 0),  loadB(wsH + SIG1, ntile, 64, 0),  acc);
        acc = wmma_f16(loadA(Hs, rowBase, 64, 32), loadB(wsH + SIG1, ntile, 64, 32), acc);
        #pragma unroll
        for (int r = 0; r < 8; ++r) {
            int m = r + half * 8;
            float v = acc[r];
            H2s[(rowBase + m) * 64 + ntile * 16 + nIdx] = (f16)(v > 0.f ? v : 0.f);
        }
    }
    // ---- sigma layer 2: H2 @ SIG2^T -> [row][16] (11 valid) ----
    {
        v8f acc = {};
        acc = wmma_f16(loadA(H2s, rowBase, 64, 0),  loadB(wsH + SIG2, 0, 64, 0),  acc);
        acc = wmma_f16(loadA(H2s, rowBase, 64, 32), loadB(wsH + SIG2, 0, 64, 32), acc);
        #pragma unroll
        for (int r = 0; r < 8; ++r) {
            int p = blockBase + rowBase + r + half * 8;
            if (p < N && nIdx < 11) {
                float v = acc[r];
                if (nIdx < 3) v *= 0.01f;
                out[p * 13 + nIdx] = v;
            }
        }
    }
}

// ---------------------------------------------------------------------------
extern "C" void kernel_launch(void* const* d_in, const int* in_sizes, int n_in,
                              void* d_out, int out_size, void* d_ws, size_t ws_size,
                              hipStream_t stream) {
    PtrPack pp;
    for (int i = 0; i < 45; ++i) pp.q[i] = (i < n_in) ? (const float*)d_in[i] : nullptr;

    int N = in_sizes[0] / 3;
    float* wsF = (float*)d_ws;
    f16*   wsH = (f16*)((char*)d_ws + 256);

    motion_prep<<<1, 256, 0, stream>>>(pp, wsF, wsH);

    int blocks = (N + 127) / 128;
    motion_main<<<blocks, 256, 0, stream>>>(
        (const float*)d_in[0],
        (const float*)d_in[3], (const float*)d_in[4], (const float*)d_in[5],
        wsF, wsH, (float*)d_out, N);
}
